// DecoderRNN_1941325218378
// MI455X (gfx1250) — compile-verified, matrix operand order checked by reading, not run on hardware
//
#include <hip/hip_runtime.h>
#include <hip/hip_bf16.h>
#include <stdint.h>

// Problem constants (from the reference)
#define V_   32000
#define E_   512
#define H_   512
#define L_   64
#define T_   64
#define B_   64
#define SOS_ 1
#define G3H  (3 * H_)   // 1536

// ---------------- WMMA bf16 infrastructure (CDNA5, wave32) ----------------
typedef __attribute__((ext_vector_type(16))) __bf16 bf16x16;
typedef __attribute__((ext_vector_type(8)))  float  floatx8;

union Frag { bf16x16 v; uint32_t u[8]; };

__device__ __forceinline__ uint16_t cvt_bf16(float f) {
  uint32_t u = __float_as_uint(f);
  u += 0x7FFFu + ((u >> 16) & 1u);   // round-to-nearest-even
  return (uint16_t)(u >> 16);
}

// Load one 16x32 bf16 fragment. Caller passes p16 = base + row*ld + k0 + kgrp
// where kgrp = (lane<16 ? 0 : 8). ISA layout: VGPR j holds K pair (kgrp+2j),
// VGPRs 4..7 hold the same pattern at K+16. Pairs are contiguous bf16 in
// row-major memory -> aligned dword loads (compiler merges to b128 pairs).
__device__ __forceinline__ void load_frag(Frag& f, const uint16_t* p16) {
  const uint32_t* p = (const uint32_t*)p16;
#pragma unroll
  for (int j = 0; j < 4; ++j) { f.u[j] = p[j]; f.u[4 + j] = p[8 + j]; }
}

__device__ __forceinline__ floatx8 wmma_bf16(const Frag& a, const Frag& b, floatx8 c) {
  return __builtin_amdgcn_wmma_f32_16x16x32_bf16(false, a.v, false, b.v,
                                                 (short)0, c, false, false);
}

// ---------------- Prep kernels (run once per call, deterministic) ----------------
__global__ void k_cvt_bf16(const float* __restrict__ src, uint16_t* __restrict__ dst, int n) {
  for (int i = blockIdx.x * blockDim.x + threadIdx.x; i < n; i += gridDim.x * blockDim.x)
    dst[i] = cvt_bf16(src[i]);
}

__global__ void k_embed(const int* __restrict__ targets, const float* __restrict__ table,
                        uint16_t* __restrict__ embs) {
  int total = T_ * B_ * E_;
  for (int i = blockIdx.x * blockDim.x + threadIdx.x; i < total; i += gridDim.x * blockDim.x) {
    int t = i / (B_ * E_);
    int r = i % (B_ * E_);
    int b = r / E_, e = r % E_;
    int id = (t == 0) ? SOS_ : targets[(t - 1) * B_ + b];
    embs[i] = cvt_bf16(table[(size_t)id * E_ + e]);
  }
}

__global__ void k_hinit(const float* __restrict__ hidden, float* __restrict__ h32,
                        uint16_t* __restrict__ hbf) {
  int i = blockIdx.x * blockDim.x + threadIdx.x;
  if (i < B_ * H_) { float v = hidden[i]; h32[i] = v; hbf[i] = cvt_bf16(v); }
}

// ---------------- Step kernels ----------------
// Attention logits [B,L] = cat(h,emb) @ W_attn.T, then rowwise softmax. 1 WG, 16 waves.
__global__ void k_attn(const uint16_t* __restrict__ hbf,
                       const uint16_t* __restrict__ emb_t,   // [B,E] for this step
                       const uint16_t* __restrict__ Wattn,   // [L, H+E] bf16
                       float* __restrict__ aw) {
  __shared__ float s_att[B_ * L_];
  int wave = threadIdx.x >> 5, lane = threadIdx.x & 31;
  int mi = wave >> 2, ni = wave & 3;
  int lm = lane & 15, kgrp = (lane < 16) ? 0 : 8;
  int m = mi * 16 + lm;   // batch row (A)
  int n = ni * 16 + lm;   // L row of W_attn (B, column-major load)
  floatx8 c = {};
  Frag a, b;
  for (int k0 = 0; k0 < H_ + E_; k0 += 32) {
    const uint16_t* arow = (k0 < H_) ? (hbf + m * H_ + k0 + kgrp)
                                     : (emb_t + m * E_ + (k0 - H_) + kgrp);
    load_frag(a, arow);
    load_frag(b, Wattn + n * (H_ + E_) + k0 + kgrp);
    c = wmma_bf16(a, b, c);
  }
  int mbase = (lane < 16) ? 0 : 8;
#pragma unroll
  for (int i = 0; i < 8; ++i)
    s_att[(mi * 16 + mbase + i) * L_ + ni * 16 + lm] = c[i];
  __syncthreads();
  if (threadIdx.x < B_) {             // one thread per batch row: softmax in f32
    int bb = threadIdx.x;
    float mx = -3.4e38f;
    for (int l = 0; l < L_; ++l) mx = fmaxf(mx, s_att[bb * L_ + l]);
    float s = 0.f;
    for (int l = 0; l < L_; ++l) s += expf(s_att[bb * L_ + l] - mx);
    float inv = 1.f / s;
    for (int l = 0; l < L_; ++l) aw[bb * L_ + l] = expf(s_att[bb * L_ + l] - mx) * inv;
  }
}

// ctx[b,h] = sum_l aw[b,l] * enc[l,b,h]  (enc fits in L2). grid = B_ blocks x 128.
__global__ void k_ctx(const float* __restrict__ aw, const float* __restrict__ enc,
                      uint16_t* __restrict__ ctxbf) {
  __shared__ float s_aw[L_];
  int b = blockIdx.x;
  if (threadIdx.x < L_) s_aw[threadIdx.x] = aw[b * L_ + threadIdx.x];
  __syncthreads();
#pragma unroll
  for (int h0 = 0; h0 < H_; h0 += 128) {
    int h = h0 + threadIdx.x;
    float acc = 0.f;
#pragma unroll 4
    for (int l = 0; l < L_; ++l)
      acc += s_aw[l] * enc[(size_t)l * B_ * H_ + (size_t)b * H_ + h];
    ctxbf[b * H_ + h] = cvt_bf16(acc);
  }
}

// rnn_in = relu(cat(emb, ctx) @ W_combine.T). grid = 32 blocks x 128 (4 waves each).
__global__ void k_combine(const uint16_t* __restrict__ emb_t,
                          const uint16_t* __restrict__ ctxbf,
                          const uint16_t* __restrict__ Wc,   // [E, E+H] bf16
                          uint16_t* __restrict__ rnnbf) {
  int wave = threadIdx.x >> 5, lane = threadIdx.x & 31;
  int mi = wave, ni = blockIdx.x;
  int lm = lane & 15, kgrp = (lane < 16) ? 0 : 8;
  int m = mi * 16 + lm, n = ni * 16 + lm;
  floatx8 c = {};
  Frag a, b;
  for (int k0 = 0; k0 < E_ + H_; k0 += 32) {
    const uint16_t* arow = (k0 < E_) ? (emb_t + m * E_ + k0 + kgrp)
                                     : (ctxbf + m * H_ + (k0 - E_) + kgrp);
    load_frag(a, arow);
    load_frag(b, Wc + n * (E_ + H_) + k0 + kgrp);
    c = wmma_bf16(a, b, c);
  }
  int mbase = (lane < 16) ? 0 : 8;
#pragma unroll
  for (int i = 0; i < 8; ++i)
    rnnbf[(mi * 16 + mbase + i) * E_ + ni * 16 + lm] = cvt_bf16(fmaxf(c[i], 0.f));
}

// gi = rnn_in@W_ih.T + b_ih ; gh = h@W_hh.T + b_hh. Fused as one N=3072 virtual GEMM.
// grid = 96 blocks x 256 (768 waves = 192 ntiles x 4 mtiles).
__global__ void k_gru_gemm(const uint16_t* __restrict__ rnnbf,
                           const uint16_t* __restrict__ hbf,
                           const uint16_t* __restrict__ Wih,  // [3H, E] bf16
                           const uint16_t* __restrict__ Whh,  // [3H, H] bf16
                           const float* __restrict__ b_ih,
                           const float* __restrict__ b_hh,
                           float* __restrict__ gi, float* __restrict__ gh) {
  int wave = (blockIdx.x * blockDim.x + threadIdx.x) >> 5;   // 0..767
  int lane = threadIdx.x & 31;
  int mi = wave & 3, nt = wave >> 2;                         // nt 0..191
  bool is_gh = nt >= 96;
  int ntile = is_gh ? nt - 96 : nt;
  const uint16_t* A  = is_gh ? hbf : rnnbf;                  // both [B,512]
  const uint16_t* W  = is_gh ? Whh : Wih;                    // [1536,512]
  const float* bia   = is_gh ? b_hh : b_ih;
  float* out         = is_gh ? gh : gi;
  int lm = lane & 15, kgrp = (lane < 16) ? 0 : 8;
  int m = mi * 16 + lm, n = ntile * 16 + lm;
  floatx8 c = {};
  Frag a, b;
  for (int k0 = 0; k0 < 512; k0 += 32) {
    load_frag(a, A + m * 512 + k0 + kgrp);
    load_frag(b, W + n * 512 + k0 + kgrp);
    c = wmma_bf16(a, b, c);
  }
  int mbase = (lane < 16) ? 0 : 8;
  int col = ntile * 16 + lm;
  float bias = bia[col];
#pragma unroll
  for (int i = 0; i < 8; ++i)
    out[(mi * 16 + mbase + i) * G3H + col] = c[i] + bias;
}

// GRU gate math in f32; updates h32 and hbf in place.
__global__ void k_gates(const float* __restrict__ gi, const float* __restrict__ gh,
                        float* __restrict__ h32, uint16_t* __restrict__ hbf) {
  int i = blockIdx.x * blockDim.x + threadIdx.x;
  if (i >= B_ * H_) return;
  int b = i >> 9, hcol = i & (H_ - 1);
  const float* gib = gi + b * G3H;
  const float* ghb = gh + b * G3H;
  float r = 1.f / (1.f + expf(-(gib[hcol] + ghb[hcol])));
  float z = 1.f / (1.f + expf(-(gib[H_ + hcol] + ghb[H_ + hcol])));
  float n = tanhf(gib[2 * H_ + hcol] + r * ghb[2 * H_ + hcol]);
  float hnew = (1.f - z) * n + z * h32[i];
  h32[i] = hnew;
  hbf[i] = cvt_bf16(hnew);
}

// logits[b,v] = h @ W_hat.T + b_hat. 2000 wave-tiles, each: 16 cols x full M=64.
// W_hat (bf16, 32.8MB) stays L2-resident across steps and streams through the
// vmem path (with global_prefetch_b8); the 64KB A matrix (h_bf) is staged ONCE
// per workgroup into LDS via gfx1250 async global->LDS DMA (ASYNCcnt) so the
// 2000 wave-tiles don't each re-pull A through WGP$/L2.
__global__ void k_logits(const uint16_t* __restrict__ hbf,
                         const uint16_t* __restrict__ What,  // [V,H] bf16
                         const float* __restrict__ b_hat,
                         float* __restrict__ logits) {
  __shared__ uint16_t s_a[B_ * H_];   // 64 KB of the 320 KB/WGP LDS

  // Async DMA h_bf (64KB) into LDS: 256 threads x 16 iters x b128.
  {
    int tid = threadIdx.x;
#pragma unroll
    for (int it = 0; it < 16; ++it) {
      int idx = (it * 256 + tid) * 8;                       // 8 bf16 = 16B chunks
      uint32_t lds_off = (uint32_t)(uintptr_t)(&s_a[idx]);  // low 32b = LDS offset
      const uint16_t* src = hbf + idx;
      asm volatile("global_load_async_to_lds_b128 %0, %1, off"
                   :: "v"(lds_off), "v"(src) : "memory");
    }
    asm volatile("s_wait_asynccnt 0x0" ::: "memory");
    __syncthreads();
  }

  int wave = (blockIdx.x * blockDim.x + threadIdx.x) >> 5;   // 0..1999
  int lane = threadIdx.x & 31;
  int lm = lane & 15, kgrp = (lane < 16) ? 0 : 8;
  int n0 = wave * 16;
  const uint16_t* wrow = What + (size_t)(n0 + lm) * H_ + kgrp;
  floatx8 c0 = {}, c1 = {}, c2 = {}, c3 = {};
  Frag a0, a1, a2, a3, b;
  for (int k0 = 0; k0 < H_; k0 += 32) {
    __builtin_prefetch((const void*)(wrow + k0 + 128), 0, 1);  // global_prefetch_b8
    load_frag(b, wrow + k0);
    load_frag(a0, s_a + (0 * 16 + lm) * H_ + k0 + kgrp);       // ds_load path
    load_frag(a1, s_a + (1 * 16 + lm) * H_ + k0 + kgrp);
    load_frag(a2, s_a + (2 * 16 + lm) * H_ + k0 + kgrp);
    load_frag(a3, s_a + (3 * 16 + lm) * H_ + k0 + kgrp);
    c0 = wmma_bf16(a0, b, c0);
    c1 = wmma_bf16(a1, b, c1);
    c2 = wmma_bf16(a2, b, c2);
    c3 = wmma_bf16(a3, b, c3);
  }
  int col = n0 + lm;
  float bias = b_hat[col];
  int mbase = (lane < 16) ? 0 : 8;
#pragma unroll
  for (int i = 0; i < 8; ++i) {
    logits[(size_t)(0  + mbase + i) * V_ + col] = c0[i] + bias;
    logits[(size_t)(16 + mbase + i) * V_ + col] = c1[i] + bias;
    logits[(size_t)(32 + mbase + i) * V_ + col] = c2[i] + bias;
    logits[(size_t)(48 + mbase + i) * V_ + col] = c3[i] + bias;
  }
}

// Per-row log-softmax over V=32000; writes final output slice for this step.
__global__ void k_logsoftmax(const float* __restrict__ logits, float* __restrict__ out_t) {
  __shared__ float red[256];
  int b = blockIdx.x, tid = threadIdx.x;
  const float* row = logits + (size_t)b * V_;
  float mx = -3.4e38f;
  for (int v = tid; v < V_; v += 256) mx = fmaxf(mx, row[v]);
  red[tid] = mx; __syncthreads();
  for (int s = 128; s > 0; s >>= 1) {
    if (tid < s) red[tid] = fmaxf(red[tid], red[tid + s]);
    __syncthreads();
  }
  mx = red[0]; __syncthreads();
  float sum = 0.f;
  for (int v = tid; v < V_; v += 256) sum += expf(row[v] - mx);
  red[tid] = sum; __syncthreads();
  for (int s = 128; s > 0; s >>= 1) {
    if (tid < s) red[tid] += red[tid + s];
    __syncthreads();
  }
  float lse = mx + logf(red[0]);
  float* orow = out_t + (size_t)b * V_;
  for (int v = tid; v < V_; v += 256) orow[v] = row[v] - lse;
}

// ---------------- Host launcher ----------------
extern "C" void kernel_launch(void* const* d_in, const int* in_sizes, int n_in,
                              void* d_out, int out_size, void* d_ws, size_t ws_size,
                              hipStream_t stream) {
  const float* hidden  = (const float*)d_in[0];
  const float* enc     = (const float*)d_in[1];
  const int*   targets = (const int*)d_in[2];
  const float* table   = (const float*)d_in[3];
  const float* W_attn  = (const float*)d_in[4];
  const float* W_comb  = (const float*)d_in[5];
  const float* W_ih    = (const float*)d_in[6];
  const float* W_hh    = (const float*)d_in[7];
  const float* b_ih    = (const float*)d_in[8];
  const float* b_hh    = (const float*)d_in[9];
  const float* W_hat   = (const float*)d_in[10];
  const float* b_hat   = (const float*)d_in[11];
  float* out = (float*)d_out;
  (void)in_sizes; (void)n_in; (void)out_size; (void)ws_size;

  // Workspace carve (~51 MB total, 256B aligned slabs)
  char* base = (char*)d_ws;
  size_t off = 0;
  auto carve = [&](size_t bytes) -> char* {
    char* p = base + off;
    off += (bytes + 255) & ~(size_t)255;
    return p;
  };
  uint16_t* Whatbf  = (uint16_t*)carve((size_t)V_ * H_ * 2);
  uint16_t* Wihbf   = (uint16_t*)carve((size_t)G3H * E_ * 2);
  uint16_t* Whhbf   = (uint16_t*)carve((size_t)G3H * H_ * 2);
  uint16_t* Wcbf    = (uint16_t*)carve((size_t)E_ * (E_ + H_) * 2);
  uint16_t* Wattnbf = (uint16_t*)carve((size_t)L_ * (H_ + E_) * 2);
  uint16_t* embsbf  = (uint16_t*)carve((size_t)T_ * B_ * E_ * 2);
  float*    h32     = (float*)carve((size_t)B_ * H_ * 4);
  uint16_t* hbf     = (uint16_t*)carve((size_t)B_ * H_ * 2);
  float*    aw      = (float*)carve((size_t)B_ * L_ * 4);
  uint16_t* ctxbf   = (uint16_t*)carve((size_t)B_ * H_ * 2);
  uint16_t* rnnbf   = (uint16_t*)carve((size_t)B_ * E_ * 2);
  float*    gi      = (float*)carve((size_t)B_ * G3H * 4);
  float*    gh      = (float*)carve((size_t)B_ * G3H * 4);
  float*    logits  = (float*)carve((size_t)B_ * V_ * 4);

  // One-time (per call) weight/bf16 prep — recomputed every call, no cached state.
  k_cvt_bf16<<<512, 256, 0, stream>>>(W_hat,  Whatbf,  V_ * H_);
  k_cvt_bf16<<<256, 256, 0, stream>>>(W_ih,   Wihbf,   G3H * E_);
  k_cvt_bf16<<<256, 256, 0, stream>>>(W_hh,   Whhbf,   G3H * H_);
  k_cvt_bf16<<<128, 256, 0, stream>>>(W_comb, Wcbf,    E_ * (E_ + H_));
  k_cvt_bf16<<<32,  256, 0, stream>>>(W_attn, Wattnbf, L_ * (H_ + E_));
  k_embed<<<512, 256, 0, stream>>>(targets, table, embsbf);
  k_hinit<<<(B_ * H_ + 255) / 256, 256, 0, stream>>>(hidden, h32, hbf);

  // Serial recurrence: stream ordering enforces step dependencies.
  for (int t = 0; t < T_; ++t) {
    const uint16_t* emb_t = embsbf + (size_t)t * B_ * E_;
    k_attn<<<1, 512, 0, stream>>>(hbf, emb_t, Wattnbf, aw);
    k_ctx<<<B_, 128, 0, stream>>>(aw, enc, ctxbf);
    k_combine<<<32, 128, 0, stream>>>(emb_t, ctxbf, Wcbf, rnnbf);
    k_gru_gemm<<<96, 256, 0, stream>>>(rnnbf, hbf, Wihbf, Whhbf, b_ih, b_hh, gi, gh);
    k_gates<<<(B_ * H_ + 255) / 256, 256, 0, stream>>>(gi, gh, h32, hbf);
    k_logits<<<250, 256, 0, stream>>>(hbf, Whatbf, b_hat, logits);
    k_logsoftmax<<<B_, 256, 0, stream>>>(logits, out + (size_t)t * B_ * V_);
  }
}
